// Attention_18760417149067
// MI455X (gfx1250) — compile-verified, matrix operand order checked by reading, not run on hardware
//
#include <hip/hip_runtime.h>

// ---------------------------------------------------------------------------
// Causal sigmoid-attention for MI455X (gfx1250), wave32 + WMMA bf16 + TDM.
//   K1: Q/K = sigmoid(X @ W) row-major bf16; V = sigmoid(X @ Wk) TRANSPOSED
//       (Vt[b][n][j]).  Outputs bounce through LDS -> coalesced b128.
//   K2: E = exp(mask(Q.K^T / sqrt(512))) unnormalized (bf16) + f32 rowsums.
//   K3: O = (E @ Vt^T) * rcp(rowsum); E/V tiles staged by the Tensor Data
//       Mover (tensor_load_to_lds, TENSORcnt), double-buffered.
// Scores bounded in [0, 22.63] (sigmoid outputs in (0,1)) -> exp() f32-safe
// without max subtraction -> single-pass softmax, normalize late.
// Block tile 64(M) x 128(N); 8 waves, each 16x64.
// ---------------------------------------------------------------------------

typedef __bf16 bf16;
typedef __attribute__((ext_vector_type(16))) __bf16 v16bf;
typedef __attribute__((ext_vector_type(8)))  __bf16 v8bf;
typedef __attribute__((ext_vector_type(8)))  float  v8f;
typedef unsigned int u32x4 __attribute__((ext_vector_type(4)));
typedef unsigned int u32x8 __attribute__((ext_vector_type(8)));

#define S_LEN  2048
#define DIM_IN 512
#define UNITS  512
#define BATCH  8
#define NROWS  (BATCH * S_LEN)   // 16384

__device__ __forceinline__ v8f v8f_zero() {
  v8f z;
#pragma unroll
  for (int i = 0; i < 8; ++i) z[i] = 0.0f;
  return z;
}

__device__ __forceinline__ float fast_sigmoid(float x) {
  return __builtin_amdgcn_rcpf(1.0f + __expf(-x));   // v_rcp_f32, no div chain
}

// 16x32 bf16 fragment (A or B of v_wmma_f32_16x16x32_bf16) from an LDS tile,
// row stride 'stride' elements (multiple of 8).  CDNA5 layout: lane&15 ->
// matrix row; lane>>4 -> K half; two b128 loads per lane.
__device__ __forceinline__ v16bf load_frag(const bf16* base, int stride,
                                           int row, int koff, int lane) {
  const int kb = (lane >> 4) << 3;                   // 0 or 8
  const bf16* p = base + row * stride + koff + kb;
  union { v16bf v; v8bf h[2]; } u;
  u.h[0] = *reinterpret_cast<const v8bf*>(p);        // K = kb + 0..7
  u.h[1] = *reinterpret_cast<const v8bf*>(p + 16);   // K = kb + 16..23
  return u.v;
}

#define WMMA_BF16(A, B, C) \
  __builtin_amdgcn_wmma_f32_16x16x32_bf16(false, (A), false, (B), (short)0, (C), false, false)

// ---------------------------------------------------------------------------
// Tensor Data Mover: 2-D bf16 tile load, Global -> LDS (ISA 08_async_tensor
// 7.x/8.x).  D# group0: count=1 | lds_addr | global_addr | type=2.
// Group1: data_size=2B, pad_enable, pad_interval=32 DWORDs (one 64-elem row),
// pad_amount=4 DWORDs (8 elems) -> LDS rows land with stride 72 elements,
// matching load_frag.  tensor dims set large (tile always in-bounds);
// tensor_dim0_stride = row stride in elements.  2-D: VADDR2/3 omitted (NULL).
// ---------------------------------------------------------------------------
__device__ __forceinline__ void tdm_load_2d(const bf16* gsrc, unsigned lds_off,
                                            unsigned tile_d0, unsigned tile_d1,
                                            unsigned row_stride_elems) {
  unsigned long long ga = (unsigned long long)(size_t)gsrc;
  u32x4 g0;
  g0[0] = 1u;                                        // count=1 (valid user D#)
  g0[1] = lds_off;                                   // lds_addr (bytes)
  g0[2] = (unsigned)ga;                              // global_addr[31:0]
  g0[3] = (unsigned)(ga >> 32) | (2u << 30);         // addr[56:32] | type=2
  u32x8 g1;
  g1[0] = (1u << 16) |                               // data_size: 2 bytes
          (1u << 20) |                               // pad_enable
          (4u << 22) |                               // pad_interval: 32 DWORDs
          (3u << 25);                                // pad_amount: 4 DWORDs
  g1[1] = 0u;                                        // dim0 lo16 = 0 (dim0=1<<20)
  g1[2] = 0x10u;                                     // dim0 hi16; dim1 lo16=0
  g1[3] = 0x10u | (tile_d0 << 16);                   // dim1 hi16 | tile_dim0
  g1[4] = tile_d1;                                   // tile_dim1 (tile_dim2=0)
  g1[5] = row_stride_elems;                          // dim0_stride lo32
  g1[6] = 0u;                                        // dim0_stride hi16 | ...
  g1[7] = 0u;
  asm volatile("tensor_load_to_lds %0, %1" :: "s"(g0), "s"(g1) : "memory");
}

// ---------------------------------------------------------------------------
// K1: 64x128 tile of Y = sigmoid(X @ W).  blockIdx.z: 0->Q, 1->K (row-major),
// 2->V transposed into Vt[b*512 + n][j].  32-deep K rounds, ping-pong LDS.
// ---------------------------------------------------------------------------
__global__ __launch_bounds__(256) void proj_sigmoid_kernel(
    const float* __restrict__ X,
    const float* __restrict__ Wq, const float* __restrict__ Wv,
    const float* __restrict__ Wk,
    bf16* __restrict__ Qd, bf16* __restrict__ Kd, bf16* __restrict__ Vt) {
  __shared__ __align__(16) bf16 SB[2][64 * 40 + 128 * 40];
  __shared__ __align__(16) bf16 Ts[64 * 136];   // output bounce tile

  const int t    = threadIdx.x;
  const int lane = t & 31;
  const int wave = t >> 5;
  const int msub = wave & 3;
  const int nsub = wave >> 2;
  const int lane15 = lane & 15;

  const int rowbase = blockIdx.x * 64;
  const int nbase   = blockIdx.y * 128;
  const int p       = blockIdx.z;
  const float* Wg = (p == 0) ? Wq : (p == 1) ? Wv : Wk;

  v8f acc[4] = {v8f_zero(), v8f_zero(), v8f_zero(), v8f_zero()};

  const int xrow = t >> 2;             // 0..63
  const int xcol = (t & 3) * 8;        // 0,8,16,24
  const int wn   = t & 127;            // 0..127
  const int wkh  = (t >> 7) * 2;       // 0 or 2

  float4 xr0, xr1;
  float  wr[16];

  auto LOAD = [&](int kk) {
    const float* xs = &X[(size_t)(rowbase + xrow) * DIM_IN + kk + xcol];
    xr0 = *(const float4*)xs;
    xr1 = *(const float4*)(xs + 4);
    if (kk + 32 < DIM_IN) __builtin_prefetch(xs + 32, 0, 1);
#pragma unroll
    for (int pp = 0; pp < 8; ++pp) {
      int k0 = pp * 4 + wkh;
      wr[2 * pp]     = Wg[(size_t)(kk + k0) * UNITS + nbase + wn];
      wr[2 * pp + 1] = Wg[(size_t)(kk + k0 + 1) * UNITS + nbase + wn];
    }
  };
  auto STORE = [&](int bufi) {
    bf16* Xs = SB[bufi];
    bf16* Ws = SB[bufi] + 64 * 40;
    union { uint4 u; bf16 e[8]; } pk;
    pk.e[0] = (bf16)xr0.x; pk.e[1] = (bf16)xr0.y;
    pk.e[2] = (bf16)xr0.z; pk.e[3] = (bf16)xr0.w;
    pk.e[4] = (bf16)xr1.x; pk.e[5] = (bf16)xr1.y;
    pk.e[6] = (bf16)xr1.z; pk.e[7] = (bf16)xr1.w;
    *(uint4*)&Xs[xrow * 40 + xcol] = pk.u;
#pragma unroll
    for (int pp = 0; pp < 8; ++pp) {
      int k0 = pp * 4 + wkh;
      union { unsigned u; bf16 e[2]; } pw;
      pw.e[0] = (bf16)wr[2 * pp]; pw.e[1] = (bf16)wr[2 * pp + 1];
      *(unsigned*)&Ws[wn * 40 + k0] = pw.u;
    }
  };

  LOAD(0); STORE(0); __syncthreads();
  const int NR = DIM_IN / 32;          // 16 rounds
  for (int r = 0; r < NR; ++r) {
    if (r + 1 < NR) LOAD((r + 1) * 32);
    const bf16* Xs = SB[r & 1];
    const bf16* Ws = SB[r & 1] + 64 * 40;
    v16bf a = load_frag(Xs, 40, msub * 16 + lane15, 0, lane);
    v16bf bm[4];
#pragma unroll
    for (int jj = 0; jj < 4; ++jj)
      bm[jj] = load_frag(Ws, 40, nsub * 64 + jj * 16 + lane15, 0, lane);
#pragma unroll
    for (int jj = 0; jj < 4; ++jj)
      acc[jj] = WMMA_BF16(a, bm[jj], acc[jj]);
    if (r + 1 < NR) STORE((r + 1) & 1);
    __syncthreads();
  }

  const int mrow = msub * 16 + ((lane >> 4) << 3);
#pragma unroll
  for (int jj = 0; jj < 4; ++jj) {
    const int ncol = nsub * 64 + jj * 16 + lane15;
#pragma unroll
    for (int r = 0; r < 8; ++r)
      Ts[(mrow + r) * 136 + ncol] = (bf16)fast_sigmoid(acc[jj][r]);
  }
  __syncthreads();
  if (p < 2) {                                  // Q / K: row-major, coalesced
    bf16* Yg = (p == 0) ? Qd : Kd;
#pragma unroll
    for (int pp = 0; pp < 4; ++pp) {
      int idx = pp * 256 + t;
      int row = idx >> 4, c = (idx & 15) * 8;
      *(uint4*)&Yg[(size_t)(rowbase + row) * UNITS + nbase + c] =
          *(uint4*)&Ts[row * 136 + c];
    }
  } else {                                      // V: transposed, coalesced
    const int b     = rowbase >> 11;
    const int jbase = rowbase & 2047;
#pragma unroll
    for (int pp = 0; pp < 4; ++pp) {
      int w = pp * 256 + t;
      int n = w >> 3;
      int c = (w & 7) * 8;
      union { uint4 u; bf16 e[8]; } pk;
#pragma unroll
      for (int u = 0; u < 8; ++u) pk.e[u] = Ts[(c + u) * 136 + n];
      *(uint4*)&Vt[((size_t)(b * UNITS) + nbase + n) * S_LEN + jbase + c] = pk.u;
    }
  }
}

// ---------------------------------------------------------------------------
// K2: E[i,j] = exp((q_i.k_j)/sqrt(512)) for j < i else 0; rowsum[i] = sum E.
// Block per (batch, 64-query tile); key tiles 128 wide, jt <= (qbase+62)>>7.
// 32-deep K rounds, ping-pong LDS, register-prefetch pipeline.
// ---------------------------------------------------------------------------
__global__ __launch_bounds__(256) void score_exp_kernel(
    const bf16* __restrict__ Qd, const bf16* __restrict__ Kd,
    bf16* __restrict__ Ed, float* __restrict__ rowsum) {
  __shared__ __align__(16) bf16 SB[2][64 * 40 + 128 * 40];
  __shared__ __align__(16) bf16 Eb[64 * 136];
  __shared__ float rsum_s[64];

  const int t    = threadIdx.x;
  const int lane = t & 31;
  const int wave = t >> 5;
  const int msub = wave & 3;
  const int nsub = wave >> 2;
  const int lane15 = lane & 15;

  const int qt    = blockIdx.x;
  const int b     = blockIdx.y;
  const int qbase = qt * 64;
  const size_t qrow0 = (size_t)b * S_LEN + qbase;

  if (t < 64) rsum_s[t] = 0.0f;
  __syncthreads();

  v8f rs = v8f_zero();
  const int mrow = msub * 16 + ((lane >> 4) << 3);
  const float scale = 0.04419417382415922f;     // 1/sqrt(512)
  const int jtmax = (qbase + 62) >> 7;

  const int qrow = t >> 2, qc = (t & 3) * 8;
  uint4 qreg, kreg[2];

  auto LOAD = [&](int jb, int kk) {
    qreg = *(const uint4*)&Qd[(qrow0 + qrow) * UNITS + kk + qc];
#pragma unroll
    for (int pp = 0; pp < 2; ++pp) {
      int idx = pp * 256 + t;
      int row = idx >> 2, c = (idx & 3) * 8;
      kreg[pp] = *(const uint4*)&Kd[((size_t)b * S_LEN + jb + row) * UNITS + kk + c];
    }
  };
  auto STORE = [&](int bufi) {
    bf16* Qs = SB[bufi];
    bf16* Ks = SB[bufi] + 64 * 40;
    *(uint4*)&Qs[qrow * 40 + qc] = qreg;
#pragma unroll
    for (int pp = 0; pp < 2; ++pp) {
      int idx = pp * 256 + t;
      int row = idx >> 2, c = (idx & 3) * 8;
      *(uint4*)&Ks[row * 40 + c] = kreg[pp];
    }
  };

  for (int jt = 0; jt <= jtmax; ++jt) {
    const int jb = jt * 128;
    v8f acc[4] = {v8f_zero(), v8f_zero(), v8f_zero(), v8f_zero()};
    LOAD(jb, 0); STORE(0); __syncthreads();
    const int NR = UNITS / 32;                  // 16 rounds
    for (int r = 0; r < NR; ++r) {
      if (r + 1 < NR) LOAD(jb, (r + 1) * 32);
      const bf16* Qs = SB[r & 1];
      const bf16* Ks = SB[r & 1] + 64 * 40;
      v16bf a = load_frag(Qs, 40, msub * 16 + lane15, 0, lane);
      v16bf bm[4];
#pragma unroll
      for (int jj = 0; jj < 4; ++jj)
        bm[jj] = load_frag(Ks, 40, nsub * 64 + jj * 16 + lane15, 0, lane);
#pragma unroll
      for (int jj = 0; jj < 4; ++jj)
        acc[jj] = WMMA_BF16(a, bm[jj], acc[jj]);
      if (r + 1 < NR) STORE((r + 1) & 1);
      __syncthreads();
    }
#pragma unroll
    for (int jj = 0; jj < 4; ++jj) {
      const int ncol = nsub * 64 + jj * 16 + lane15;
      const int jcol = jb + ncol;
#pragma unroll
      for (int r = 0; r < 8; ++r) {
        const int i = qbase + mrow + r;
        float e = (jcol < i) ? __expf(acc[jj][r] * scale) : 0.0f;
        Eb[(mrow + r) * 136 + ncol] = (bf16)e;
        rs[r] += e;
      }
    }
    __syncthreads();
#pragma unroll
    for (int pp = 0; pp < 4; ++pp) {
      int idx = pp * 256 + t;
      int row = idx >> 4, c = (idx & 15) * 8;
      *(uint4*)&Ed[((size_t)b * S_LEN + qbase + row) * S_LEN + jb + c] =
          *(uint4*)&Eb[row * 136 + c];
    }
  }
#pragma unroll
  for (int r = 0; r < 8; ++r)
    atomicAdd(&rsum_s[mrow + r], rs[r]);
  __syncthreads();
  if (t < 64) rowsum[qrow0 + t] = rsum_s[t];
}

// ---------------------------------------------------------------------------
// K3: O = (E @ V) * rcp(rowsum).  Block per (qtile, 128-wide ntile, batch);
// 64-wide j chunks, ping-pong LDS.  Tile staging is done entirely by the
// Tensor Data Mover: wave 0 issues the E-tile D#, wave 1 the V-tile D#,
// one chunk ahead; issuing waves s_wait_tensorcnt(0) before the barrier.
// ---------------------------------------------------------------------------
__global__ __launch_bounds__(256) void attn_out_kernel(
    const bf16* __restrict__ Ed, const bf16* __restrict__ Vt,
    const float* __restrict__ rowsum, float* __restrict__ Out) {
  // per buffer: Es 64x64 (stride 72) then Vs 128x64 (stride 72, [n][j])
  __shared__ __align__(16) bf16 SB[2][64 * 72 + 128 * 72];

  const int t    = threadIdx.x;
  const int lane = t & 31;
  const int wave = t >> 5;
  const int msub = wave & 3;
  const int nsub = wave >> 2;
  const int lane15 = lane & 15;

  const int qt    = blockIdx.x;
  const int nt    = blockIdx.y;
  const int b     = blockIdx.z;
  const int qbase = qt * 64;
  const int nbase = nt * 128;

  v8f acc[4] = {v8f_zero(), v8f_zero(), v8f_zero(), v8f_zero()};
  const int jcmax = (((qbase + 62) >> 7) << 1) + 1;   // 64-wide chunks

  const bf16* Erow = &Ed[((size_t)b * S_LEN + qbase) * S_LEN];
  const bf16* Vrow = &Vt[((size_t)(b * UNITS) + nbase) * S_LEN];

  auto ISSUE = [&](int bufi, int jb) {          // TDM: stage chunk jb -> bufi
    if (wave == 0)
      tdm_load_2d(Erow + jb, (unsigned)(size_t)&SB[bufi][0],
                  /*tile_d0=*/64, /*tile_d1=*/64, /*stride=*/S_LEN);
    else if (wave == 1)
      tdm_load_2d(Vrow + jb, (unsigned)(size_t)&SB[bufi][64 * 72],
                  /*tile_d0=*/64, /*tile_d1=*/128, /*stride=*/S_LEN);
  };

  ISSUE(0, 0);
  if (wave < 2) __builtin_amdgcn_s_wait_tensorcnt(0);
  __syncthreads();
  for (int jc = 0; jc <= jcmax; ++jc) {
    if (jc < jcmax) ISSUE((jc + 1) & 1, (jc + 1) * 64);
    const bf16* Es = SB[jc & 1];
    const bf16* Vs = SB[jc & 1] + 64 * 72;
#pragma unroll
    for (int ks = 0; ks < 64; ks += 32) {
      v16bf a = load_frag(Es, 72, msub * 16 + lane15, ks, lane);
      v16bf bm[4];
#pragma unroll
      for (int jj = 0; jj < 4; ++jj)
        bm[jj] = load_frag(Vs, 72, nsub * 64 + jj * 16 + lane15, ks, lane);
#pragma unroll
      for (int jj = 0; jj < 4; ++jj)
        acc[jj] = WMMA_BF16(a, bm[jj], acc[jj]);
    }
    if (wave < 2) __builtin_amdgcn_s_wait_tensorcnt(0);
    __syncthreads();
  }

  const int mrow = msub * 16 + ((lane >> 4) << 3);
#pragma unroll
  for (int r = 0; r < 8; ++r) {
    const int i = qbase + mrow + r;
    float rsv = rowsum[(size_t)b * S_LEN + i];
    float inv = (rsv > 0.0f) ? __builtin_amdgcn_rcpf(rsv) : 0.0f;
    size_t orow = ((size_t)b * S_LEN + i) * UNITS;
#pragma unroll
    for (int jj = 0; jj < 4; ++jj)
      Out[orow + nbase + nsub * 64 + jj * 16 + lane15] = acc[jj][r] * inv;
  }
}

// ---------------------------------------------------------------------------
// Launch.  Workspace (bytes): Q,K,Vt bf16 3*16MB + E bf16 64MB + rowsum 64KB.
// ---------------------------------------------------------------------------
extern "C" void kernel_launch(void* const* d_in, const int* in_sizes, int n_in,
                              void* d_out, int out_size, void* d_ws,
                              size_t ws_size, hipStream_t stream) {
  const float* X  = (const float*)d_in[0];
  const float* Wq = (const float*)d_in[1];
  const float* Wv = (const float*)d_in[2];
  const float* Wk = (const float*)d_in[3];

  bf16* Qd = (bf16*)d_ws;
  bf16* Kd = Qd + (size_t)NROWS * UNITS;
  bf16* Vt = Kd + (size_t)NROWS * UNITS;        // transposed: [b*512+n][j]
  bf16* Ed = Vt + (size_t)NROWS * UNITS;
  float* rowsum = (float*)(Ed + (size_t)BATCH * S_LEN * S_LEN);
  float* Out = (float*)d_out;

  proj_sigmoid_kernel<<<dim3(NROWS / 64, UNITS / 128, 3), 256, 0, stream>>>(
      X, Wq, Wv, Wk, Qd, Kd, Vt);
  score_exp_kernel<<<dim3(S_LEN / 64, BATCH), 256, 0, stream>>>(
      Qd, Kd, Ed, rowsum);
  attn_out_kernel<<<dim3(S_LEN / 64, UNITS / 128, BATCH), 256, 0, stream>>>(
      Ed, Vt, rowsum, Out);
}
